// MultiHeadAttention_61615600829001
// MI455X (gfx1250) — matrix operand order repaired and checked by reading, not compiled, not run on hardware
//
#include <hip/hip_runtime.h>
#include <hip/hip_bf16.h>
#include <math.h>

// ---------------------------------------------------------------------------
// CDNA5 (gfx1250) multi-head attention, bf16 WMMA pipeline with TDM staging.
// b=4, n=2048, d_in=d_out=1024, heads=16, head_dim=64.
// ---------------------------------------------------------------------------

typedef __bf16  v16bf __attribute__((ext_vector_type(16)));
typedef float   v8f   __attribute__((ext_vector_type(8)));
typedef unsigned int u32x4 __attribute__((ext_vector_type(4)));
typedef unsigned int u32x8 __attribute__((ext_vector_type(8)));

union Frag {                 // one WMMA A or B operand (16 bf16 = 8 VGPRs)
    v16bf v;
    u32x4 q[2];
};

__device__ __forceinline__ v8f wmma_bf16(const Frag& a, const Frag& b, v8f c) {
    return __builtin_amdgcn_wmma_f32_16x16x32_bf16(
        /*neg_a=*/false, a.v, /*neg_b=*/false, b.v,
        /*c_mod=*/(short)0, c, /*reuse_a=*/false, /*reuse_b=*/false);
}

#define BATCH 4
#define SEQ   2048
#define DMODEL 1024
#define HEADS 16
#define HDIM  64
#define MROWS (BATCH*SEQ)      // 8192

// ---------------------------------------------------------------------------
// Tensor Data Mover: issue a 128x32-bf16 tile load (global -> LDS).
// D# per CDNA5 ISA. 2D tensor, data_size=2B,
// tile 32 elems x 128 rows, LDS rows padded 64B->80B (pad_interval=16 DW,
// pad_amount=4 DW) for 16B-aligned, bank-conflict-free ds_load_b128.
// ---------------------------------------------------------------------------
#define TROW_B   80                      // staged bytes per row
#define TILE_B   (128 * TROW_B)          // 10240 bytes per tile
#define STAGE_B  (2 * TILE_B)            // A tile + B tile per stage

__device__ __forceinline__ u32x8 tdm_group1(unsigned K) {
    u32x8 g1;
    // [15:0] wg_mask=0 | [17:16] data_size=1(2B) | [20] pad_enable
    // [24:22] pad_interval=3 (16 DW) | [31:25] pad_amount=3 (4 DW)
    g1[0] = (1u << 16) | (1u << 20) | (3u << 22) | (3u << 25);
    g1[1] = (32u << 16);                 // [79:48] tensor_dim0 = 32 (low 16)
    g1[2] = (128u << 16);                // [95:80] tensor_dim1 = 128 (low 16)
    g1[3] = (32u << 16);                 // [127:112] tile_dim0 = 32
    g1[4] = 128u;                        // [143:128] tile_dim1 = 128
    g1[5] = K;                           // [207:160] tensor_dim0_stride = K elems
    g1[6] = 0u;
    g1[7] = 0u;
    return g1;
}

__device__ __forceinline__ void tdm_load_tile(unsigned lds_off, const __bf16* gptr,
                                              u32x8 g1) {
    unsigned long long ga = (unsigned long long)(__UINTPTR_TYPE__)gptr;
    u32x4 g0;
    g0[0] = 1u;                                          // count=1 (valid D#)
    g0[1] = lds_off;                                     // [63:32] lds_addr
    g0[2] = (unsigned)ga;                                // [95:64] global_addr lo
    g0[3] = (unsigned)((ga >> 32) & 0x1FFFFFFu) | (2u << 30);  // addr hi | type=2
    asm volatile("tensor_load_to_lds %0, %1" :: "s"(g0), "s"(g1) : "memory");
}

// ---------------------------------------------------------------------------
// fp32 -> bf16 streaming convert
// ---------------------------------------------------------------------------
__global__ __launch_bounds__(256)
void cvt_bf16(const float* __restrict__ src, __bf16* __restrict__ dst, int n) {
    int i = blockIdx.x * 256 + threadIdx.x;
    if (i < n) dst[i] = (__bf16)src[i];
}

// fp32 (K x N) -> bf16 transposed (N x K)
__global__ __launch_bounds__(256)
void cvtT_bf16(const float* __restrict__ src, __bf16* __restrict__ dst, int K, int N) {
    int i = blockIdx.x * 256 + threadIdx.x;
    if (i < K * N) {
        int k = i / N, n = i % N;
        dst[(size_t)n * K + k] = (__bf16)src[i];
    }
}

// ---------------------------------------------------------------------------
// bf16 GEMM with TDM-staged, double-buffered LDS tiles:
//   C(MxN) = A(MxK, row-major) * BT(NxK, row-major)^T
// Workgroup = 256 threads = 8 waves; WG tile 128x128; wave tile 32x64.
// Wave 0 issues two tensor_load_to_lds per 32-K stage; all waves
// s_wait_tensorcnt + barrier, then feed WMMA from conflict-free LDS.
// mode 0: store bf16 to (b,h,n,hd)        (Q, K projections)
// mode 1: store bf16 to (b,h,hd,n)        (V projection, pre-transposed)
// mode 2: store fp32 row-major + bias     (output projection)
// ---------------------------------------------------------------------------
__global__ __launch_bounds__(256)
void gemm_bf16(const __bf16* __restrict__ A, const __bf16* __restrict__ BT,
               int M, int N, int K, int mode,
               __bf16* __restrict__ outb, float* __restrict__ outf,
               const float* __restrict__ bias) {
    __shared__ __align__(16) unsigned char smem[2 * STAGE_B];   // 40 KB

    const int wid  = threadIdx.x >> 5;
    const int lane = threadIdx.x & 31;
    const int lh   = lane >> 4;      // which 16-lane half
    const int ll   = lane & 15;

    const int mblk = blockIdx.x * 128;
    const int nblk = blockIdx.y * 128;
    const int wm   = (wid & 3) * 32;   // wave M base within WG tile
    const int wn   = (wid >> 2) * 64;  // wave N base within WG tile

    const unsigned smem_base = (unsigned)(__UINTPTR_TYPE__)&smem[0];
    const u32x8 g1 = tdm_group1((unsigned)K);

    const int nk = K / 32;
    v8f acc[2][4] = {};

    // prologue: stage 0
    if (wid == 0) {
        tdm_load_tile(smem_base,          A  + (size_t)mblk * K, g1);
        tdm_load_tile(smem_base + TILE_B, BT + (size_t)nblk * K, g1);
    }

    for (int kt = 0; kt < nk; ++kt) {
        __builtin_amdgcn_s_wait_tensorcnt(0);   // no-op for non-issuing waves
        __syncthreads();                        // stage kt resident in LDS

        if (kt + 1 < nk && wid == 0) {
            unsigned nb = smem_base + ((kt + 1) & 1) * STAGE_B;
            tdm_load_tile(nb,          A  + (size_t)mblk * K + (kt + 1) * 32, g1);
            tdm_load_tile(nb + TILE_B, BT + (size_t)nblk * K + (kt + 1) * 32, g1);
        }

        const unsigned char* sA = smem + (kt & 1) * STAGE_B;
        const unsigned char* sB = sA + TILE_B;

        Frag a[2], b[4];
        // A frag (16x32): lanes 0-15 K{0..7,16..23}, lanes 16-31 K{8..15,24..31}
#pragma unroll
        for (int i = 0; i < 2; ++i) {
            const unsigned char* p = sA + (wm + i * 16 + ll) * TROW_B + 16 * lh;
            a[i].q[0] = *reinterpret_cast<const u32x4*>(p);
            a[i].q[1] = *reinterpret_cast<const u32x4*>(p + 32);
        }
        // B frag (32x16): lane = col N; lanes 0-15 K{0..7,16..23}, lanes 16-31 K{8..15,24..31}
#pragma unroll
        for (int j = 0; j < 4; ++j) {
            const unsigned char* p = sB + (wn + j * 16 + ll) * TROW_B + 16 * lh;
            b[j].q[0] = *reinterpret_cast<const u32x4*>(p);
            b[j].q[1] = *reinterpret_cast<const u32x4*>(p + 32);
        }
#pragma unroll
        for (int i = 0; i < 2; ++i)
#pragma unroll
            for (int j = 0; j < 4; ++j)
                acc[i][j] = wmma_bf16(a[i], b[j], acc[i][j]);

        // LDS reads must land before TDM may overwrite this buffer (kt+2)
        asm volatile("s_wait_dscnt 0x0" ::: "memory");
        __syncthreads();
    }

    // Epilogue. C layout: VGPR r, lanes 0-15 -> M=r, lanes 16-31 -> M=r+8; N = lane%16.
#pragma unroll
    for (int i = 0; i < 2; ++i)
#pragma unroll
        for (int j = 0; j < 4; ++j)
#pragma unroll
            for (int r = 0; r < 8; ++r) {
                int m = mblk + wm + i * 16 + r + 8 * lh;
                int n = nblk + wn + j * 16 + ll;
                float v = acc[i][j][r];
                if (mode == 2) {
                    outf[(size_t)m * N + n] = v + bias[n];
                } else {
                    int bi = m >> 11, ni = m & 2047;   // SEQ = 2048
                    int hi = n >> 6,  di = n & 63;     // HDIM = 64
                    size_t off = (mode == 0)
                        ? (((size_t)(bi * HEADS + hi) * SEQ + ni) * HDIM + di)
                        : (((size_t)(bi * HEADS + hi) * HDIM + di) * SEQ + ni);
                    outb[off] = (__bf16)v;
                }
            }
}

// ---------------------------------------------------------------------------
// Causal flash attention. Q,K: (b,h,n,hd) bf16; VT: (b,h,hd,n) bf16.
// Grid: (b*h, n/128); block 256 = 8 waves; each wave owns 16 query rows.
// Online softmax over 32-key chunks; P re-layout via per-wave LDS scratch.
// ---------------------------------------------------------------------------
__global__ __launch_bounds__(256)
void attn_flash(const __bf16* __restrict__ Q, const __bf16* __restrict__ Kb,
                const __bf16* __restrict__ VT, __bf16* __restrict__ ctx) {
    __shared__ __bf16 pscr[8][16 * 32];   // per-wave 16x32 bf16 P tile (8 KB)

    const int wid  = threadIdx.x >> 5;
    const int lane = threadIdx.x & 31;
    const int lh   = lane >> 4;
    const int ll   = lane & 15;

    const int bh = blockIdx.x;            // 0..63
    const int bi = bh >> 4, hi = bh & 15;
    const int q0 = blockIdx.y * 128 + wid * 16;

    const __bf16* Qh = Q  + (size_t)bh * SEQ * HDIM;
    const __bf16* Kh = Kb + (size_t)bh * SEQ * HDIM;
    const __bf16* Vh = VT + (size_t)bh * HDIM * SEQ;

    // Q fragments for this wave's 16 rows (reused across all key chunks)
    Frag aq[2];
#pragma unroll
    for (int ks = 0; ks < 2; ++ks) {
        const __bf16* p = Qh + (size_t)(q0 + ll) * HDIM + ks * 32 + 8 * lh;
        aq[ks].q[0] = *reinterpret_cast<const u32x4*>(p);
        aq[ks].q[1] = *reinterpret_cast<const u32x4*>(p + 16);
    }

    v8f O[4] = {};
    float mrun[8], lrun[8];
#pragma unroll
    for (int r = 0; r < 8; ++r) { mrun[r] = -INFINITY; lrun[r] = 0.0f; }

    const float c = 0.125f * 1.44269504088896f;   // 1/sqrt(64) * log2(e)
    const int kend = q0 + 16;                     // causal: keys < kend

    for (int kb = 0; kb < kend; kb += 32) {
        // ----- scores: 2 key tiles x 2 K-steps -----
        v8f S[2] = {};
#pragma unroll
        for (int t = 0; t < 2; ++t)
#pragma unroll
            for (int ks = 0; ks < 2; ++ks) {
                Frag bk;
                const __bf16* p = Kh + (size_t)(kb + t * 16 + ll) * HDIM + ks * 32 + 8 * lh;
                bk.q[0] = *reinterpret_cast<const u32x4*>(p);
                bk.q[1] = *reinterpret_cast<const u32x4*>(p + 16);
                S[t] = wmma_bf16(aq[ks], bk, S[t]);
            }

        // ----- causal mask + online softmax (rows split across lane halves) -----
        float mnew[8], alpha[8];
#pragma unroll
        for (int r = 0; r < 8; ++r) {
            int row  = q0 + r + 8 * lh;
            float s0 = (kb + ll      <= row) ? S[0][r] : -INFINITY;
            float s1 = (kb + 16 + ll <= row) ? S[1][r] : -INFINITY;
            float v = fmaxf(s0, s1);
#pragma unroll
            for (int msk = 1; msk < 16; msk <<= 1)
                v = fmaxf(v, __shfl_xor(v, msk, 32));
            mnew[r]  = fmaxf(mrun[r], v);
            alpha[r] = exp2f((mrun[r] - mnew[r]) * c);
            float p0 = exp2f((s0 - mnew[r]) * c);
            float p1 = exp2f((s1 - mnew[r]) * c);
            float rs = p0 + p1;
#pragma unroll
            for (int msk = 1; msk < 16; msk <<= 1)
                rs += __shfl_xor(rs, msk, 32);
            lrun[r] = lrun[r] * alpha[r] + rs;
            mrun[r] = mnew[r];
            // stash P in wave-private LDS, row-major 16x32
            pscr[wid][(r + 8 * lh) * 32 + ll]      = (__bf16)p0;
            pscr[wid][(r + 8 * lh) * 32 + 16 + ll] = (__bf16)p1;
        }
#pragma unroll
        for (int t2 = 0; t2 < 4; ++t2)
#pragma unroll
            for (int r = 0; r < 8; ++r)
                O[t2][r] *= alpha[r];

        // ----- P re-layout (C layout -> A-fragment layout) via LDS -----
        Frag pa;
        const __bf16* sp = &pscr[wid][ll * 32 + 8 * lh];
        pa.q[0] = *reinterpret_cast<const u32x4*>(sp);
        pa.q[1] = *reinterpret_cast<const u32x4*>(sp + 16);

        // ----- O += P @ V (V stored transposed: contiguous keys per hd row) -----
#pragma unroll
        for (int t2 = 0; t2 < 4; ++t2) {
            Frag bv;
            const __bf16* p = Vh + (size_t)(t2 * 16 + ll) * SEQ + kb + 8 * lh;
            bv.q[0] = *reinterpret_cast<const u32x4*>(p);
            bv.q[1] = *reinterpret_cast<const u32x4*>(p + 16);
            O[t2] = wmma_bf16(pa, bv, O[t2]);
        }
    }

    // ----- epilogue: ctx (b, n, h*hd) bf16 -----
#pragma unroll
    for (int t2 = 0; t2 < 4; ++t2)
#pragma unroll
        for (int r = 0; r < 8; ++r) {
            int row = q0 + r + 8 * lh;
            int col = hi * HDIM + t2 * 16 + ll;
            ctx[(size_t)(bi * SEQ + row) * DMODEL + col] = (__bf16)(O[t2][r] / lrun[r]);
        }
}

// ---------------------------------------------------------------------------
// Host-side orchestration
// ---------------------------------------------------------------------------
extern "C" void kernel_launch(void* const* d_in, const int* in_sizes, int n_in,
                              void* d_out, int out_size, void* d_ws, size_t ws_size,
                              hipStream_t stream) {
    const float* x  = (const float*)d_in[0];
    const float* Wq = (const float*)d_in[1];
    const float* Wk = (const float*)d_in[2];
    const float* Wv = (const float*)d_in[3];
    const float* Wo = (const float*)d_in[4];
    const float* bo = (const float*)d_in[5];
    float* out = (float*)d_out;

    __bf16* ws = (__bf16*)d_ws;
    const size_t NX   = (size_t)MROWS * DMODEL;       // 8388608
    const size_t NW   = (size_t)DMODEL * DMODEL;      // 1048576
    __bf16* xb   = ws;                 // [0, NX)
    __bf16* WqT  = xb  + NX;
    __bf16* WkT  = WqT + NW;
    __bf16* WvT  = WkT + NW;
    __bf16* WoT  = WvT + NW;
    __bf16* Qb   = WoT + NW;           // (b,h,n,hd)
    __bf16* Kbuf = Qb  + NX;           // (b,h,n,hd)
    __bf16* VTb  = Kbuf + NX;          // (b,h,hd,n)
    __bf16* ctxb = VTb + NX;           // (b*n, d)

    // 1) precision conversion (+ weight transpose for contiguous B fragments)
    cvt_bf16<<<(int)((NX + 255) / 256), 256, 0, stream>>>(x, xb, (int)NX);
    cvtT_bf16<<<(int)((NW + 255) / 256), 256, 0, stream>>>(Wq, WqT, DMODEL, DMODEL);
    cvtT_bf16<<<(int)((NW + 255) / 256), 256, 0, stream>>>(Wk, WkT, DMODEL, DMODEL);
    cvtT_bf16<<<(int)((NW + 255) / 256), 256, 0, stream>>>(Wv, WvT, DMODEL, DMODEL);
    cvtT_bf16<<<(int)((NW + 255) / 256), 256, 0, stream>>>(Wo, WoT, DMODEL, DMODEL);

    // 2) Q/K/V projections (M=8192, N=1024, K=1024)
    dim3 ggrid(MROWS / 128, DMODEL / 128);
    gemm_bf16<<<ggrid, 256, 0, stream>>>(xb, WqT, MROWS, DMODEL, DMODEL, 0, Qb,   nullptr, nullptr);
    gemm_bf16<<<ggrid, 256, 0, stream>>>(xb, WkT, MROWS, DMODEL, DMODEL, 0, Kbuf, nullptr, nullptr);
    gemm_bf16<<<ggrid, 256, 0, stream>>>(xb, WvT, MROWS, DMODEL, DMODEL, 1, VTb,  nullptr, nullptr);

    // 3) causal flash attention
    attn_flash<<<dim3(BATCH * HEADS, SEQ / 128), 256, 0, stream>>>(Qb, Kbuf, VTb, ctxb);

    // 4) output projection + bias -> fp32
    gemm_bf16<<<ggrid, 256, 0, stream>>>(ctxb, WoT, MROWS, DMODEL, DMODEL, 2, nullptr, out, bo);
}
